// Swin3D_67362267070724
// MI455X (gfx1250) — matrix operand order repaired and checked by reading, not compiled, not run on hardware
//
#include <hip/hip_runtime.h>
#include <hip/hip_bf16.h>
#include <math.h>

#define D_MODEL 128
#define N_HEADS 8
#define LN_EPS 1e-5f

typedef __bf16 v16bf __attribute__((ext_vector_type(16)));
typedef float  v8f   __attribute__((ext_vector_type(8)));

// ---------------------------------------------------------------------------
// Weight prep: split fp32 W[K x N] into bf16 hi/lo, stored in WMMA B-fragment
// order: index = (((t * nK + ks) * 32 + lane) * 16 + i), where per CDNA5 ISA
// 7.12.2 element i of lane's fragment is B[k][n] with
//   k = ks*32 + (lane>>4)*8 + (i&7) + (i>>3)*16,   n = t*16 + (lane&15).
// 3 bf16 WMMAs (hi*hi + hi*lo + lo*hi) then reproduce ~fp32 accuracy.
// ---------------------------------------------------------------------------
__global__ void __launch_bounds__(256)
split_weight_kernel(const float* __restrict__ W, __bf16* __restrict__ hi,
                    __bf16* __restrict__ lo, int K, int N) {
  const int idx = blockIdx.x * 256 + threadIdx.x;
  if (idx >= K * N) return;
  const int i    = idx & 15;
  const int lane = (idx >> 4) & 31;
  const int rest = idx >> 9;
  const int nK   = K >> 5;
  const int ks   = rest % nK;
  const int t    = rest / nK;
  const int v = i >> 1, p = i & 1;
  const int k = (ks << 5) + ((lane >> 4) << 3) + (i & 7) + ((i >> 3) << 4);
  const int n = (t << 4) + (lane & 15);
  const float  x = W[(size_t)k * N + n];
  const __bf16 h = (__bf16)x;
  hi[idx] = h;
  lo[idx] = (__bf16)(x - (float)h);
}

// ---------------------------------------------------------------------------
// A fragment (16x32 tile of fp32 activations, row-major, lda elements).
// Lane layout (ISA 7.12.2): m = lane&15, half = lane>>4; the 16 elements are
// two contiguous 8-float runs: k0+half*8+{0..7} and k0+16+half*8+{0..7}
// -> 4 x 16B vector loads, then split to bf16 hi/lo.
// ---------------------------------------------------------------------------
__device__ __forceinline__ void load_a_frag(const float* __restrict__ A, int lda,
                                            int k0, int lane, v16bf& hi, v16bf& lo) {
  const int m    = lane & 15;
  const int half = lane >> 4;
  const float* row = A + (size_t)m * lda + k0 + (half << 3);
  float f[16];
  *(float4*)(f + 0)  = *(const float4*)(row + 0);
  *(float4*)(f + 4)  = *(const float4*)(row + 4);
  *(float4*)(f + 8)  = *(const float4*)(row + 16);
  *(float4*)(f + 12) = *(const float4*)(row + 20);
#pragma unroll
  for (int i = 0; i < 16; ++i) {
    const __bf16 h = (__bf16)f[i];
    hi[i] = h;
    lo[i] = (__bf16)(f[i] - (float)h);
  }
}

#define WMMA_BF16(A_, B_, C_) \
  __builtin_amdgcn_wmma_f32_16x16x32_bf16(false, (A_), false, (B_), (short)0, (C_), false, false)

// ---------------------------------------------------------------------------
// GEMM: C[M x Ncols] = A[M x K] @ B[K x Ncols] (+bias) (+relu)
// One wave owns a 16-row x 128-col tile (8 accumulators); A fragment loaded
// once per K-step and reused across 8 column tiles -> 24 WMMA per K-step.
// grid = (ceil(M/16/8), Ncols/128), block = 256 (8 independent waves).
// ---------------------------------------------------------------------------
__global__ void __launch_bounds__(256)
gemm_wmma_kernel(const float* __restrict__ A,
                 const __bf16* __restrict__ Bhi, const __bf16* __restrict__ Blo,
                 const float* __restrict__ bias, float* __restrict__ C,
                 int M, int K, int Ncols, int relu) {
  const int wave = threadIdx.x >> 5;
  const int lane = threadIdx.x & 31;
  const int tile = blockIdx.x * 8 + wave;
  if ((tile << 4) >= M) return;                 // uniform per wave
  const int row0   = tile << 4;
  const int colblk = blockIdx.y << 3;           // base col-tile (128 cols per y)
  const int nK     = K >> 5;

  v8f acc[8];
#pragma unroll
  for (int t = 0; t < 8; ++t) acc[t] = (v8f){};

  for (int ks = 0; ks < nK; ++ks) {
    v16bf ahi, alo;
    load_a_frag(A + (size_t)row0 * K, K, ks << 5, lane, ahi, alo);
#pragma unroll
    for (int t = 0; t < 8; ++t) {
      const size_t fidx = ((size_t)((colblk + t) * nK + ks) * 32 + lane) << 4;
      const v16bf bhi = *(const v16bf*)(Bhi + fidx);
      const v16bf blo = *(const v16bf*)(Blo + fidx);
      acc[t] = WMMA_BF16(ahi, bhi, acc[t]);
      acc[t] = WMMA_BF16(ahi, blo, acc[t]);
      acc[t] = WMMA_BF16(alo, bhi, acc[t]);
    }
  }

  const int n = lane & 15, half = lane >> 4;
#pragma unroll
  for (int t = 0; t < 8; ++t) {
    const int col = ((colblk + t) << 4) + n;
    const float bv = bias ? bias[col] : 0.0f;
#pragma unroll
    for (int i = 0; i < 8; ++i) {
      const int r = row0 + i + (half << 3);
      float v = acc[t][i] + bv;
      if (relu) v = fmaxf(v, 0.0f);
      C[(size_t)r * Ncols + col] = v;
    }
  }
}

// ---------------------------------------------------------------------------
// Fused GEMM + bias + residual + LayerNorm (Ncols fixed = 128):
// Out = LN( R + A @ B + bias ) * g + b
// The wave holds complete 16 rows in registers, so the row mean/variance is a
// per-lane tile sum + a 16-lane shfl_xor tree. No LDS, no __syncthreads.
// ---------------------------------------------------------------------------
__global__ void __launch_bounds__(256)
gemm_ln_wmma_kernel(const float* __restrict__ A,
                    const __bf16* __restrict__ Bhi, const __bf16* __restrict__ Blo,
                    const float* __restrict__ bias, const float* __restrict__ R,
                    const float* __restrict__ g, const float* __restrict__ bln,
                    float* __restrict__ Out, int M, int K) {
  const int wave = threadIdx.x >> 5;
  const int lane = threadIdx.x & 31;
  const int tile = blockIdx.x * 8 + wave;
  if ((tile << 4) >= M) return;
  const int row0 = tile << 4;
  const int nK   = K >> 5;

  v8f acc[8];
#pragma unroll
  for (int t = 0; t < 8; ++t) acc[t] = (v8f){};

  for (int ks = 0; ks < nK; ++ks) {
    v16bf ahi, alo;
    load_a_frag(A + (size_t)row0 * K, K, ks << 5, lane, ahi, alo);
#pragma unroll
    for (int t = 0; t < 8; ++t) {
      const size_t fidx = ((size_t)(t * nK + ks) * 32 + lane) << 4;
      const v16bf bhi = *(const v16bf*)(Bhi + fidx);
      const v16bf blo = *(const v16bf*)(Blo + fidx);
      acc[t] = WMMA_BF16(ahi, bhi, acc[t]);
      acc[t] = WMMA_BF16(ahi, blo, acc[t]);
      acc[t] = WMMA_BF16(alo, bhi, acc[t]);
    }
  }

  const int n = lane & 15, half = lane >> 4;
  float s[8], ss[8];
#pragma unroll
  for (int i = 0; i < 8; ++i) { s[i] = 0.0f; ss[i] = 0.0f; }

  // bias + residual, accumulate row statistics
#pragma unroll
  for (int t = 0; t < 8; ++t) {
    const int col = (t << 4) + n;
    const float bv = bias[col];
#pragma unroll
    for (int i = 0; i < 8; ++i) {
      const int r = row0 + i + (half << 3);
      const float v = acc[t][i] + bv + R[(size_t)r * D_MODEL + col];
      acc[t][i] = v;
      s[i] += v;
      ss[i] += v * v;
    }
  }
  // reduce across the 16 lanes of this half-wave (rows 0-7 in lanes 0-15,
  // rows 8-15 in lanes 16-31 -> independent groups)
#pragma unroll
  for (int i = 0; i < 8; ++i) {
#pragma unroll
    for (int mk = 1; mk < 16; mk <<= 1) {
      s[i]  += __shfl_xor(s[i], mk, 16);
      ss[i] += __shfl_xor(ss[i], mk, 16);
    }
  }
  float mu[8], rs[8];
#pragma unroll
  for (int i = 0; i < 8; ++i) {
    mu[i] = s[i] * (1.0f / D_MODEL);
    const float var = ss[i] * (1.0f / D_MODEL) - mu[i] * mu[i];
    rs[i] = rsqrtf(var + LN_EPS);
  }
#pragma unroll
  for (int t = 0; t < 8; ++t) {
    const int col = (t << 4) + n;
    const float gc = g[col], bc = bln[col];
#pragma unroll
    for (int i = 0; i < 8; ++i) {
      const int r = row0 + i + (half << 3);
      Out[(size_t)r * D_MODEL + col] = (acc[t][i] - mu[i]) * rs[i] * gc + bc;
    }
  }
}

// ---------------------------------------------------------------------------
// Edge attention: 128 threads per edge (coalesced 512B row gathers).
// Each 16-lane group = one head: shfl-xor dot reduce, score = exp(clip(./4)),
// atomic scatter of V*score into wV and score into z.
// ---------------------------------------------------------------------------
__global__ void __launch_bounds__(256)
edge_attn_kernel(const int* __restrict__ src, const int* __restrict__ dst,
                 const float* __restrict__ Q, const float* __restrict__ Km,
                 const float* __restrict__ V, float* __restrict__ wV,
                 float* __restrict__ z, int E) {
  const int t = blockIdx.x * 256 + threadIdx.x;
  const int e = t >> 7;
  if (e >= E) return;
  const int d = t & (D_MODEL - 1);

  const int s  = src[e];
  const int dt = dst[e];

  const float kv = Km[(size_t)s  * D_MODEL + d];
  const float qv = Q [(size_t)dt * D_MODEL + d];
  float prod = kv * qv;
  prod += __shfl_xor(prod, 1, 16);
  prod += __shfl_xor(prod, 2, 16);
  prod += __shfl_xor(prod, 4, 16);
  prod += __shfl_xor(prod, 8, 16);

  const float scaled = fminf(fmaxf(prod * 0.25f, -5.0f), 5.0f); // /sqrt(16)
  const float score  = expf(scaled);

  atomicAdd(&wV[(size_t)dt * D_MODEL + d], V[(size_t)s * D_MODEL + d] * score);
  if ((d & 15) == 0)
    atomicAdd(&z[(size_t)dt * N_HEADS + (d >> 4)], score);
}

// head_out = where(z > 0, wV / z, wV)  (in place on wV)
__global__ void __launch_bounds__(256)
finalize_heads_kernel(float* __restrict__ wV, const float* __restrict__ z, int total) {
  const int i = blockIdx.x * 256 + threadIdx.x;
  if (i >= total) return;
  const int node = i >> 7;
  const int d    = i & (D_MODEL - 1);
  const float zv = z[(size_t)node * N_HEADS + (d >> 4)];
  if (zv > 0.0f) wV[i] /= zv;
}

// ---------------------------------------------------------------------------
extern "C" void kernel_launch(void* const* d_in, const int* in_sizes, int n_in,
                              void* d_out, int out_size, void* d_ws, size_t ws_size,
                              hipStream_t stream) {
  const float* h   = (const float*)d_in[0];
  const int*   src = (const int*)  d_in[1];
  const int*   dst = (const int*)  d_in[2];
  const float* Wq  = (const float*)d_in[3];
  const float* Wk  = (const float*)d_in[4];
  const float* Wv  = (const float*)d_in[5];
  const float* Wo  = (const float*)d_in[6];
  const float* bo  = (const float*)d_in[7];
  const float* g1  = (const float*)d_in[8];
  const float* b1  = (const float*)d_in[9];
  const float* g2  = (const float*)d_in[10];
  const float* b2  = (const float*)d_in[11];
  const float* W1  = (const float*)d_in[12];
  const float* c1  = (const float*)d_in[13];
  const float* W2  = (const float*)d_in[14];
  const float* c2  = (const float*)d_in[15];
  float* out = (float*)d_out;

  const int N = in_sizes[0] / D_MODEL;   // 100000 (multiple of 16)
  const int E = in_sizes[1];             // 1600000

  // ---- workspace layout ----
  float* ws  = (float*)d_ws;
  float* Q   = ws;
  float* K   = Q   + (size_t)N * D_MODEL;
  float* V   = K   + (size_t)N * D_MODEL;
  float* wV  = V   + (size_t)N * D_MODEL;
  float* z   = wV  + (size_t)N * D_MODEL;
  float* hh  = z   + (size_t)N * N_HEADS;
  float* mid = hh  + (size_t)N * D_MODEL;           // N x 256
  __bf16* wb = (__bf16*)(mid + (size_t)N * 2 * D_MODEL);
  const int SZ_DD = D_MODEL * D_MODEL;              // 16384
  const int SZ_D2 = D_MODEL * 2 * D_MODEL;          // 32768
  __bf16* Wq_hi = wb;                __bf16* Wq_lo = Wq_hi + SZ_DD;
  __bf16* Wk_hi = Wq_lo + SZ_DD;     __bf16* Wk_lo = Wk_hi + SZ_DD;
  __bf16* Wv_hi = Wk_lo + SZ_DD;     __bf16* Wv_lo = Wv_hi + SZ_DD;
  __bf16* Wo_hi = Wv_lo + SZ_DD;     __bf16* Wo_lo = Wo_hi + SZ_DD;
  __bf16* W1_hi = Wo_lo + SZ_DD;     __bf16* W1_lo = W1_hi + SZ_D2;
  __bf16* W2_hi = W1_lo + SZ_D2;     __bf16* W2_lo = W2_hi + SZ_D2;

  hipMemsetAsync(wV, 0, (size_t)N * D_MODEL * sizeof(float), stream);
  hipMemsetAsync(z,  0, (size_t)N * N_HEADS * sizeof(float), stream);

  const dim3 blk(256);
  const int tiles16   = (N + 15) >> 4;              // 6250
  const int gemmBlks  = (tiles16 + 7) >> 3;         // 782

  // ---- weight split/swizzle (512 KB total; L2 resident) ----
  split_weight_kernel<<<dim3((SZ_DD + 255) / 256), blk, 0, stream>>>(Wq, Wq_hi, Wq_lo, D_MODEL, D_MODEL);
  split_weight_kernel<<<dim3((SZ_DD + 255) / 256), blk, 0, stream>>>(Wk, Wk_hi, Wk_lo, D_MODEL, D_MODEL);
  split_weight_kernel<<<dim3((SZ_DD + 255) / 256), blk, 0, stream>>>(Wv, Wv_hi, Wv_lo, D_MODEL, D_MODEL);
  split_weight_kernel<<<dim3((SZ_DD + 255) / 256), blk, 0, stream>>>(Wo, Wo_hi, Wo_lo, D_MODEL, D_MODEL);
  split_weight_kernel<<<dim3((SZ_D2 + 255) / 256), blk, 0, stream>>>(W1, W1_hi, W1_lo, D_MODEL, 2 * D_MODEL);
  split_weight_kernel<<<dim3((SZ_D2 + 255) / 256), blk, 0, stream>>>(W2, W2_hi, W2_lo, 2 * D_MODEL, D_MODEL);

  // ---- Q/K/V projections ----
  gemm_wmma_kernel<<<dim3(gemmBlks, 1), blk, 0, stream>>>(h, Wq_hi, Wq_lo, nullptr, Q, N, D_MODEL, D_MODEL, 0);
  gemm_wmma_kernel<<<dim3(gemmBlks, 1), blk, 0, stream>>>(h, Wk_hi, Wk_lo, nullptr, K, N, D_MODEL, D_MODEL, 0);
  gemm_wmma_kernel<<<dim3(gemmBlks, 1), blk, 0, stream>>>(h, Wv_hi, Wv_lo, nullptr, V, N, D_MODEL, D_MODEL, 0);

  // ---- edge attention scatter (2 edges per 256-thread block) ----
  edge_attn_kernel<<<dim3((E + 1) / 2), blk, 0, stream>>>(src, dst, Q, K, V, wV, z, E);

  // ---- head_out = wV / z ----
  finalize_heads_kernel<<<dim3((N * D_MODEL + 255) / 256), blk, 0, stream>>>(wV, z, N * D_MODEL);

  // ---- hh = LN(h + head_out @ Wo + bo) ----
  gemm_ln_wmma_kernel<<<dim3(gemmBlks), blk, 0, stream>>>(wV, Wo_hi, Wo_lo, bo, h, g1, b1, hh, N, D_MODEL);

  // ---- mid = relu(hh @ W1 + c1)   (N x 256) ----
  gemm_wmma_kernel<<<dim3(gemmBlks, 2), blk, 0, stream>>>(hh, W1_hi, W1_lo, c1, mid, N, D_MODEL, 2 * D_MODEL, 1);

  // ---- out = LN(hh + mid @ W2 + c2) ----
  gemm_ln_wmma_kernel<<<dim3(gemmBlks), blk, 0, stream>>>(mid, W2_hi, W2_lo, c2, hh, g2, b2, out, N, 2 * D_MODEL);
}
